// MLP_24438363914265
// MI455X (gfx1250) — matrix operand order repaired and checked by reading, not compiled, not run on hardware
//
#include <hip/hip_runtime.h>
#include <math.h>

typedef _Float16 h4  __attribute__((ext_vector_type(4)));
typedef _Float16 h8  __attribute__((ext_vector_type(8)));
typedef _Float16 h16 __attribute__((ext_vector_type(16)));
typedef float    f4  __attribute__((ext_vector_type(4)));
typedef float    v8f __attribute__((ext_vector_type(8)));

#define ABSMEAN_EPS 1e-6f
#define LN_EPS      1e-5f

// ---------------------------------------------------------------------------
// CDNA5 async global->LDS staging (ASYNCcnt-tracked).
// Builtin signature (from hipcc diagnostic): (v4i AS1*, v4i AS3*, Ii, Ii)
// ---------------------------------------------------------------------------
typedef int v4i __attribute__((vector_size(16)));
typedef __attribute__((address_space(1))) v4i* as1_v4i_p;
typedef __attribute__((address_space(3))) v4i* as3_v4i_p;

__device__ __forceinline__ void async_copy_b128(const _Float16* g, _Float16* l) {
#if __has_builtin(__builtin_amdgcn_global_load_async_to_lds_b128)
    __builtin_amdgcn_global_load_async_to_lds_b128((as1_v4i_p)g, (as3_v4i_p)l, 0, 0);
#else
    *(h8*)l = *(const h8*)g;
#endif
}

__device__ __forceinline__ void wait_async0() {
#if __has_builtin(__builtin_amdgcn_s_wait_asynccnt)
    __builtin_amdgcn_s_wait_asynccnt(0);
#else
    asm volatile("s_wait_asynccnt 0" ::: "memory");
#endif
}

// ---------------------------------------------------------------------------
// 1) sum(|W|) reduction -> accum
// ---------------------------------------------------------------------------
__global__ __launch_bounds__(256) void absum_kernel(const float* __restrict__ W,
                                                    int n, float* __restrict__ accum) {
    __shared__ float red[256];
    float s = 0.f;
    for (int i = blockIdx.x * blockDim.x + threadIdx.x; i < n;
         i += gridDim.x * blockDim.x)
        s += fabsf(W[i]);
    red[threadIdx.x] = s;
    __syncthreads();
    for (int off = 128; off > 0; off >>= 1) {
        if ((int)threadIdx.x < off) red[threadIdx.x] += red[threadIdx.x + off];
        __syncthreads();
    }
    if (threadIdx.x == 0) atomicAdd(accum, red[0]);
}

// ---------------------------------------------------------------------------
// 2) ternary quantize to fp16 (exact): Wq = sign(W) * min(rint(|W|/gamma), 1)
// ---------------------------------------------------------------------------
__global__ __launch_bounds__(256) void quant_kernel(const float* __restrict__ W,
                                                    int n,
                                                    const float* __restrict__ accum,
                                                    _Float16* __restrict__ Wq) {
    const float gamma = accum[0] / (float)n + ABSMEAN_EPS;
    const float inv   = 1.f / gamma;
    int i = (blockIdx.x * blockDim.x + threadIdx.x) * 4;
    if (i + 3 < n) {
        f4 w = *(const f4*)(W + i);
        h4 q;
        for (int j = 0; j < 4; ++j) {
            float a = fminf(rintf(fabsf(w[j]) * inv), 1.f);
            q[j] = (_Float16)copysignf(a, w[j]);
        }
        *(h4*)(Wq + i) = q;
    }
}

// ---------------------------------------------------------------------------
// 2b) fp32 -> fp16 convert (activations), so both GEMMs stage pure fp16
// ---------------------------------------------------------------------------
__global__ __launch_bounds__(256) void cvt_kernel(const float* __restrict__ X,
                                                  _Float16* __restrict__ Y, int n) {
    int i = (blockIdx.x * blockDim.x + threadIdx.x) * 4;
    if (i + 3 < n) {
        f4 v = *(const f4*)(X + i);
        h4 hv;
        for (int j = 0; j < 4; ++j) hv[j] = (_Float16)v[j];
        *(h4*)(Y + i) = hv;
    }
}

// ---------------------------------------------------------------------------
// 3/5) WMMA GEMM: C[M,N] = A[M,K] * B[N,K]^T, fp16 inputs, fp32 accumulate.
//   Block tile 128x128, BK=32, 8 waves, wave tile 64x32 (4x2 fragments).
//   Double-buffered LDS fed by async global->LDS b128 copies.
// ---------------------------------------------------------------------------
#define BM 128
#define BN 128
#define BK 32
#define LDT 40  // LDS row stride in halves (80B: keeps b128 alignment)

template <int N, int K, bool GELU, typename OT>
__global__ __launch_bounds__(256) void wmma_gemm_kernel(const _Float16* __restrict__ A,
                                                        const _Float16* __restrict__ B,
                                                        OT* __restrict__ C) {
    __shared__ _Float16 As[2][BM * LDT];
    __shared__ _Float16 Bs[2][BN * LDT];

    const int tid  = threadIdx.x;
    const int wave = tid >> 5;
    const int lane = tid & 31;
    const int hsel = lane >> 4;  // wave half (ISA 16-bit operand layout)
    const int l16  = lane & 15;

    const int m0 = blockIdx.y * BM;
    const int n0 = blockIdx.x * BN;
    const int wm = (wave & 1) * 64;   // wave M offset: 4 fragment rows
    const int wn = (wave >> 1) * 32;  // wave N offset: 2 fragment cols

    // ---- per-thread staging slots (fixed for all K-steps) ----
    const int rS = tid >> 2;          // 0..63   (rows rS and rS+64)
    const int cS = (tid & 3) * 8;     // 0,8,16,24
    const _Float16* gA = A + (size_t)(m0 + rS) * K + cS;
    const _Float16* gB = B + (size_t)(n0 + rS) * K + cS;
    const int lOff = rS * LDT + cS;

    v8f acc[4][2];
    for (int mi = 0; mi < 4; ++mi)
        for (int ni = 0; ni < 2; ++ni)
            for (int j = 0; j < 8; ++j) acc[mi][ni][j] = 0.f;

    auto stage = [&](int buf, int kt) {
        const _Float16* ga = gA + kt * BK;
        const _Float16* gb = gB + kt * BK;
        _Float16* la = &As[buf][lOff];
        _Float16* lb = &Bs[buf][lOff];
        async_copy_b128(ga, la);
        async_copy_b128(ga + (size_t)64 * K, la + 64 * LDT);
        async_copy_b128(gb, lb);
        async_copy_b128(gb + (size_t)64 * K, lb + 64 * LDT);
    };

    union HV { h16 v; h8 h[2]; };

    constexpr int NK = K / BK;
    stage(0, 0);
    wait_async0();
    __syncthreads();

    for (int kt = 0; kt < NK; ++kt) {
        const int cur = kt & 1;
        if (kt + 1 < NK) stage(cur ^ 1, kt + 1);  // overlap copy with compute

        // ---- fragments per ISA 7.12.2 lane layout ----
        HV a[4], b[2];
#pragma unroll
        for (int mi = 0; mi < 4; ++mi) {
            const _Float16* p = &As[cur][(wm + mi * 16 + l16) * LDT + hsel * 8];
            a[mi].h[0] = *(const h8*)p;         // K = hsel*8 + 0..7
            a[mi].h[1] = *(const h8*)(p + 16);  // K = hsel*8 + 16..23
        }
#pragma unroll
        for (int ni = 0; ni < 2; ++ni) {
            const _Float16* p = &Bs[cur][(wn + ni * 16 + l16) * LDT + hsel * 8];
            b[ni].h[0] = *(const h8*)p;        // K = hsel*8 + 0..7
            b[ni].h[1] = *(const h8*)(p + 16); // K = hsel*8 + 16..23
        }
#pragma unroll
        for (int mi = 0; mi < 4; ++mi)
#pragma unroll
            for (int ni = 0; ni < 2; ++ni)
                acc[mi][ni] = __builtin_amdgcn_wmma_f32_16x16x32_f16(
                    false, a[mi].v, false, b[ni].v, (short)0, acc[mi][ni],
                    false, false);

        wait_async0();
        __syncthreads();
    }

    // ---- epilogue: C/D layout M = v + 8*hsel, N = l16 ----
#pragma unroll
    for (int mi = 0; mi < 4; ++mi)
#pragma unroll
        for (int ni = 0; ni < 2; ++ni) {
            const int col = n0 + wn + ni * 16 + l16;
#pragma unroll
            for (int v = 0; v < 8; ++v) {
                const int row = m0 + wm + mi * 16 + hsel * 8 + v;
                float xv = acc[mi][ni][v];
                if constexpr (GELU)
                    xv = 0.5f * xv * (1.f + erff(xv * 0.70710678118654752f));
                C[(size_t)row * N + col] = (OT)xv;
            }
        }
}

// ---------------------------------------------------------------------------
// 4) LayerNorm over 4096-wide rows, in place on fp16 H, fp32 math.
// ---------------------------------------------------------------------------
__global__ __launch_bounds__(256) void ln_kernel(_Float16* __restrict__ H,
                                                 const float* __restrict__ w,
                                                 const float* __restrict__ b) {
    __shared__ float s1[256], s2[256];
    const int ncols = 4096;
    _Float16* row = H + (size_t)blockIdx.x * ncols;
    float x[16];
    float sum = 0.f, sq = 0.f;
#pragma unroll
    for (int i = 0; i < 16; ++i) {
        float v = (float)row[threadIdx.x + i * 256];
        x[i] = v;
        sum += v;
        sq += v * v;
    }
    s1[threadIdx.x] = sum;
    s2[threadIdx.x] = sq;
    __syncthreads();
    for (int off = 128; off > 0; off >>= 1) {
        if ((int)threadIdx.x < off) {
            s1[threadIdx.x] += s1[threadIdx.x + off];
            s2[threadIdx.x] += s2[threadIdx.x + off];
        }
        __syncthreads();
    }
    const float mean = s1[0] / (float)ncols;
    const float var  = s2[0] / (float)ncols - mean * mean;
    const float rstd = rsqrtf(var + LN_EPS);
#pragma unroll
    for (int i = 0; i < 16; ++i) {
        int c = threadIdx.x + i * 256;
        row[c] = (_Float16)((x[i] - mean) * rstd * w[c] + b[c]);
    }
}

// ---------------------------------------------------------------------------
extern "C" void kernel_launch(void* const* d_in, const int* in_sizes, int n_in,
                              void* d_out, int out_size, void* d_ws,
                              size_t ws_size, hipStream_t stream) {
    const float* x      = (const float*)d_in[0];
    const float* W_fc   = (const float*)d_in[1];
    const float* ln_w   = (const float*)d_in[2];
    const float* ln_b   = (const float*)d_in[3];
    const float* W_proj = (const float*)d_in[4];
    float* out          = (float*)d_out;

    const int D  = 1024;
    const int Hd = 4096;
    const int M  = in_sizes[0] / D;       // 16384 tokens
    const int NW = Hd * D;                // 4,194,304 weights each
    const int NX = M * D;                 // 16,777,216 activations

    char* ws = (char*)d_ws;
    float*    sums  = (float*)ws;                                   // [2]
    _Float16* Wq_fc = (_Float16*)(ws + 256);
    _Float16* Wq_pj = (_Float16*)(ws + 256 + (size_t)NW * 2);
    _Float16* Hbuf  = (_Float16*)(ws + 256 + (size_t)NW * 4);
    _Float16* X16   = (_Float16*)(ws + 256 + (size_t)NW * 4 + (size_t)M * Hd * 2);

    (void)hipMemsetAsync(sums, 0, 256, stream);

    absum_kernel<<<1024, 256, 0, stream>>>(W_fc,   NW, &sums[0]);
    absum_kernel<<<1024, 256, 0, stream>>>(W_proj, NW, &sums[1]);

    quant_kernel<<<NW / (256 * 4), 256, 0, stream>>>(W_fc,   NW, &sums[0], Wq_fc);
    quant_kernel<<<NW / (256 * 4), 256, 0, stream>>>(W_proj, NW, &sums[1], Wq_pj);
    cvt_kernel<<<NX / (256 * 4), 256, 0, stream>>>(x, X16, NX);

    // GEMM1: (M x D) * (Hd x D)^T -> fp16 H, fused exact GELU
    dim3 g1(Hd / BN, M / BM);
    wmma_gemm_kernel<4096, 1024, true, _Float16>
        <<<g1, 256, 0, stream>>>(X16, Wq_fc, Hbuf);

    // LayerNorm over Hd, in place
    ln_kernel<<<M, 256, 0, stream>>>(Hbuf, ln_w, ln_b);

    // GEMM2: (M x Hd) * (D x Hd)^T -> fp32 out
    dim3 g2(D / BN, M / BM);
    wmma_gemm_kernel<1024, 4096, false, float>
        <<<g2, 256, 0, stream>>>(Hbuf, Wq_pj, out);
}
